// SelfAttention_29042568855951
// MI455X (gfx1250) — matrix operand order repaired and checked by reading, not compiled, not run on hardware
//
#include <hip/hip_runtime.h>
#include <hip/hip_bf16.h>

#define B_  8
#define C_  512
#define L_  2048
#define CK_ 64

typedef __attribute__((ext_vector_type(16))) __bf16 v16bf;
typedef __attribute__((ext_vector_type(8)))  float  v8f;

union BFrag {
    v16bf v;
    uint4 u[2];
};

static __device__ inline v8f wmma_bf16(const v16bf& a, const v16bf& b, const v8f& c) {
    return __builtin_amdgcn_wmma_f32_16x16x32_bf16(false, a, false, b, (short)0, c, false, false);
}

// ---------------------------------------------------------------------------
// Kernel: generic f32 -> bf16 convert (used once for Wq/Wk/Wv; L2-resident)
// ---------------------------------------------------------------------------
__global__ __launch_bounds__(256) void k_cvt(const float* __restrict__ src,
                                             __bf16* __restrict__ dst, int n) {
    int i = blockIdx.x * 256 + threadIdx.x;
    if (i < n) dst[i] = (__bf16)src[i];
}

// ---------------------------------------------------------------------------
// Kernel: x[B,C,L] f32 -> xT[B,L,C] bf16  (LDS 32x32 tile transpose)
// ---------------------------------------------------------------------------
__global__ __launch_bounds__(256) void k_transpose(const float* __restrict__ x,
                                                   __bf16* __restrict__ xT) {
    __shared__ float tile[32][33];
    const int b  = blockIdx.z;
    const int c0 = blockIdx.y * 32;
    const int l0 = blockIdx.x * 32;
    const int tx = threadIdx.x;   // 0..31
    const int ty = threadIdx.y;   // 0..7

    const float* xp = x + ((size_t)b * C_ + c0) * L_ + l0;
#pragma unroll
    for (int i = 0; i < 4; i++) {
        int c = ty + 8 * i;
        tile[c][tx] = xp[(size_t)c * L_ + tx];
    }
    __syncthreads();
    __bf16* op = xT + ((size_t)b * L_ + l0) * C_ + c0;
#pragma unroll
    for (int i = 0; i < 4; i++) {
        int l = ty + 8 * i;
        op[(size_t)l * C_ + tx] = (__bf16)tile[tx][l];
    }
}

// ---------------------------------------------------------------------------
// Kernel: projection GEMM  y[o,l] = sum_c W[o,c] * x[c,l] + bias[o]
// Each wave: 16 o x 64 l. Fully unrolled K loop with ping-pong register
// stages (constant stage index -> no buffer copies).
// ---------------------------------------------------------------------------
__global__ __launch_bounds__(128) void k_proj(const __bf16* __restrict__ Wb,
                                              const float* __restrict__ bias,
                                              const __bf16* __restrict__ xT,
                                              __bf16* __restrict__ outT,
                                              __bf16* __restrict__ outN) {
    const int b     = blockIdx.z;
    const int oBase = blockIdx.y * 16;
    const int wid   = threadIdx.x >> 5;
    const int t     = threadIdx.x & 31;
    const int g     = t >> 4;
    const int tn    = t & 15;
    const int lBase = blockIdx.x * 256 + wid * 64;

    const v8f vzero = {0.f, 0.f, 0.f, 0.f, 0.f, 0.f, 0.f, 0.f};
    v8f acc[4];
#pragma unroll
    for (int n = 0; n < 4; n++) acc[n] = vzero;

    const __bf16* wrow  = Wb + (size_t)(oBase + tn) * C_;
    const __bf16* xbase = xT + ((size_t)b * L_ + lBase + tn) * C_;

    BFrag aB[2], bB[2][4];
    auto load_stage = [&](int k, int st) {
        aB[st].u[0] = *(const uint4*)(wrow + k + 8 * g);
        aB[st].u[1] = *(const uint4*)(wrow + k + 16 + 8 * g);
#pragma unroll
        for (int n = 0; n < 4; n++) {
            const __bf16* xr = xbase + (size_t)(16 * n) * C_ + k + 8 * g;
            bB[st][n].u[0] = *(const uint4*)(xr);
            bB[st][n].u[1] = *(const uint4*)(xr + 16);
        }
    };

    load_stage(0, 0);
#pragma unroll
    for (int i = 0; i < C_ / 32; i++) {
        const int cur = i & 1;
        if (i < C_ / 32 - 1) load_stage(32 * (i + 1), cur ^ 1);
#pragma unroll
        for (int n = 0; n < 4; n++)
            acc[n] = wmma_bf16(aB[cur].v, bB[cur][n].v, acc[n]);
    }

    if (outT) {
#pragma unroll
        for (int n = 0; n < 4; n++) {
            const int l = lBase + 16 * n + tn;
#pragma unroll
            for (int j = 0; j < 8; j++) {
                int o = oBase + j + 8 * g;
                float vv = acc[n][j] + bias[o];
                outT[((size_t)b * L_ + l) * CK_ + o] = (__bf16)vv;
            }
        }
    } else {
#pragma unroll
        for (int n = 0; n < 4; n++) {
            const int l = lBase + 16 * n + tn;
#pragma unroll
            for (int j = 0; j < 8; j++) {
                int o = oBase + j + 8 * g;
                float vv = acc[n][j] + bias[o];
                outN[((size_t)b * C_ + o) * L_ + l] = (__bf16)vv;
            }
        }
    }
}

// ---------------------------------------------------------------------------
// Kernel: fused attention for one (batch, 32-query tile).
// 8 waves per workgroup. Pass 1: softmax stats, fully unrolled with
// ping-pong k stages. Pass 2: ping-pong attn LDS buffer, ONE barrier per
// 128-key chunk; next chunk's k-fragments issued before the consume loop;
// v B-fragments pipelined inside the fully-unrolled consume loop.
// ---------------------------------------------------------------------------
__global__ __launch_bounds__(256) void k_attn(const __bf16* __restrict__ qT,
                                              const __bf16* __restrict__ kT,
                                              const __bf16* __restrict__ vW,
                                              const float* __restrict__ x,
                                              const float* __restrict__ gamma,
                                              float* __restrict__ out) {
    __shared__ __bf16 attnLds[2][32][136];   // ping-pong 32 q rows x 128 keys
    __shared__ float  mW[8][32];
    __shared__ float  sW[8][32];
    __shared__ float  mG[32];
    __shared__ float  rG[32];

    const int b     = blockIdx.y;
    const int qBase = blockIdx.x * 32;
    const int tid   = threadIdx.x;
    const int w     = tid >> 5;
    const int t     = tid & 31;
    const int g     = t >> 4;
    const int tn    = t & 15;

    // q A-fragments: 2 q-subtiles x 2 K-steps, kept in registers for both passes
    BFrag aq[2][2];
#pragma unroll
    for (int qs = 0; qs < 2; qs++) {
        const __bf16* qrow = qT + ((size_t)b * L_ + qBase + 16 * qs + tn) * CK_;
#pragma unroll
        for (int s = 0; s < 2; s++) {
            aq[qs][s].u[0] = *(const uint4*)(qrow + 32 * s + 8 * g);
            aq[qs][s].u[1] = *(const uint4*)(qrow + 32 * s + 16 + 8 * g);
        }
    }

    const v8f vzero = {0.f, 0.f, 0.f, 0.f, 0.f, 0.f, 0.f, 0.f};

    auto load_k = [&](int keyBase, BFrag& b0, BFrag& b1) {
        const __bf16* krow = kT + ((size_t)b * L_ + keyBase + tn) * CK_;
        b0.u[0] = *(const uint4*)(krow + 8 * g);
        b0.u[1] = *(const uint4*)(krow + 16 + 8 * g);
        b1.u[0] = *(const uint4*)(krow + 32 + 8 * g);
        b1.u[1] = *(const uint4*)(krow + 48 + 8 * g);
    };

    // ---------------- Pass 1: row max / sum-exp ----------------
    float m[2][8], s[2][8];
#pragma unroll
    for (int qs = 0; qs < 2; qs++)
#pragma unroll
        for (int j = 0; j < 8; j++) { m[qs][j] = -INFINITY; s[qs][j] = 0.f; }

    BFrag kB[2][2];
    load_k(16 * w, kB[0][0], kB[0][1]);
#pragma unroll
    for (int i = 0; i < 16; i++) {
        const int cur = i & 1;
        if (i < 15) load_k((w + 8 * (i + 1)) * 16, kB[cur ^ 1][0], kB[cur ^ 1][1]);
#pragma unroll
        for (int qs = 0; qs < 2; qs++) {
            v8f e = vzero;
            e = wmma_bf16(aq[qs][0].v, kB[cur][0].v, e);
            e = wmma_bf16(aq[qs][1].v, kB[cur][1].v, e);
#pragma unroll
            for (int j = 0; j < 8; j++) {
                float ev = e[j];
                float mn = fmaxf(m[qs][j], ev);
                s[qs][j] = s[qs][j] * __expf(m[qs][j] - mn) + __expf(ev - mn);
                m[qs][j] = mn;
            }
        }
    }
    // reduce across the 16 lanes of each half-wave group
#pragma unroll
    for (int qs = 0; qs < 2; qs++) {
#pragma unroll
        for (int j = 0; j < 8; j++) {
            for (int off = 1; off < 16; off <<= 1) {
                float mo = __shfl_xor(m[qs][j], off, 32);
                float so = __shfl_xor(s[qs][j], off, 32);
                float mn = fmaxf(m[qs][j], mo);
                s[qs][j] = s[qs][j] * __expf(m[qs][j] - mn) + so * __expf(mo - mn);
                m[qs][j] = mn;
            }
        }
    }
    if (tn == 0) {
#pragma unroll
        for (int qs = 0; qs < 2; qs++)
#pragma unroll
            for (int j = 0; j < 8; j++) {
                mW[w][16 * qs + j + 8 * g] = m[qs][j];
                sW[w][16 * qs + j + 8 * g] = s[qs][j];
            }
    }
    __syncthreads();
    if (tid < 32) {
        float mm = -INFINITY;
        for (int ww = 0; ww < 8; ww++) mm = fmaxf(mm, mW[ww][tid]);
        float ss = 0.f;
        for (int ww = 0; ww < 8; ww++) ss += sW[ww][tid] * __expf(mW[ww][tid] - mm);
        mG[tid] = mm;
        rG[tid] = 1.0f / ss;
    }
    __syncthreads();

    float rowM[2][8], rowR[2][8];
#pragma unroll
    for (int qs = 0; qs < 2; qs++)
#pragma unroll
        for (int j = 0; j < 8; j++) {
            rowM[qs][j] = mG[16 * qs + j + 8 * g];
            rowR[qs][j] = rG[16 * qs + j + 8 * g];
        }

    // computes attn tile for this wave's 16 keys of a chunk, stores into buf
    auto attn_tile = [&](const BFrag& b0, const BFrag& b1, int buf) {
#pragma unroll
        for (int qs = 0; qs < 2; qs++) {
            v8f e = vzero;
            e = wmma_bf16(aq[qs][0].v, b0.v, e);
            e = wmma_bf16(aq[qs][1].v, b1.v, e);
#pragma unroll
            for (int j = 0; j < 8; j++)
                attnLds[buf][16 * qs + j + 8 * g][16 * w + tn] =
                    (__bf16)(__expf(e[j] - rowM[qs][j]) * rowR[qs][j]);
        }
    };

    // ---------------- Pass 2: out = attn * v^T ----------------
    v8f acc[2][4];
#pragma unroll
    for (int qs = 0; qs < 2; qs++)
#pragma unroll
        for (int cs = 0; cs < 4; cs++) acc[qs][cs] = vzero;
    const int cW = w * 64;   // this wave's 64-channel slice

    // prologue: attn slab for chunk 0 into buf 0
    BFrag kC0, kC1;
    load_k(16 * w, kC0, kC1);
    attn_tile(kC0, kC1, 0);
    __syncthreads();

    for (int chunk = 0; chunk < 16; chunk++) {
        const int buf      = chunk & 1;
        const int keyChunk = chunk * 128;

        // issue next chunk's k-fragment loads early (consumed after the GEMM)
        BFrag kN0, kN1;
        if (chunk < 15) load_k((chunk + 1) * 128 + 16 * w, kN0, kN1);

        // consume attn slab: out-GEMM over this 128-key chunk
#pragma unroll
        for (int s2 = 0; s2 < 4; s2++) {
            BFrag a0, a1;
            const __bf16* arow0 = &attnLds[buf][tn][0];
            const __bf16* arow1 = &attnLds[buf][16 + tn][0];
            a0.u[0] = *(const uint4*)(arow0 + 32 * s2 + 8 * g);
            a0.u[1] = *(const uint4*)(arow0 + 32 * s2 + 16 + 8 * g);
            a1.u[0] = *(const uint4*)(arow1 + 32 * s2 + 8 * g);
            a1.u[1] = *(const uint4*)(arow1 + 32 * s2 + 16 + 8 * g);

            const __bf16* vbase = vW + ((size_t)b * C_ + cW + tn) * L_ +
                                  keyChunk + 32 * s2 + 8 * g;
            BFrag bvC;
            bvC.u[0] = *(const uint4*)(vbase);
            bvC.u[1] = *(const uint4*)(vbase + 16);
#pragma unroll
            for (int cs = 0; cs < 4; cs++) {
                BFrag bvN;
                if (cs < 3) {
                    const __bf16* vr = vbase + (size_t)(16 * (cs + 1)) * L_;
                    bvN.u[0] = *(const uint4*)(vr);
                    bvN.u[1] = *(const uint4*)(vr + 16);
                }
                acc[0][cs] = wmma_bf16(a0.v, bvC.v, acc[0][cs]);
                acc[1][cs] = wmma_bf16(a1.v, bvC.v, acc[1][cs]);
                bvC = bvN;
            }
        }

        // produce next chunk's attn slab into the other buffer
        if (chunk < 15) attn_tile(kN0, kN1, buf ^ 1);
        __syncthreads();
    }

    // ---------------- epilogue: gamma*out + x ----------------
    const float gma = gamma[0];
#pragma unroll
    for (int qs = 0; qs < 2; qs++) {
#pragma unroll
        for (int cs = 0; cs < 4; cs++) {
            const int c = cW + 16 * cs + tn;
#pragma unroll
            for (int j = 0; j < 8; j++) {
                const int q = qBase + 16 * qs + j + 8 * g;
                const size_t idx = ((size_t)b * C_ + c) * L_ + q;
                out[idx] = gma * acc[qs][cs][j] + x[idx];
            }
        }
    }
}

// ---------------------------------------------------------------------------
extern "C" void kernel_launch(void* const* d_in, const int* in_sizes, int n_in,
                              void* d_out, int out_size, void* d_ws, size_t ws_size,
                              hipStream_t stream) {
    (void)in_sizes; (void)n_in; (void)out_size; (void)ws_size;

    const float* x     = (const float*)d_in[0];
    const float* Wq    = (const float*)d_in[1];
    const float* bq    = (const float*)d_in[2];
    const float* Wk    = (const float*)d_in[3];
    const float* bk    = (const float*)d_in[4];
    const float* Wv    = (const float*)d_in[5];
    const float* bv    = (const float*)d_in[6];
    const float* gamma = (const float*)d_in[7];
    float* out = (float*)d_out;

    // workspace layout (bf16): xT | qT | kT | v | Wq | Wk | Wv  (~38.4 MB)
    __bf16* xT  = (__bf16*)d_ws;
    __bf16* qT  = xT  + (size_t)B_ * L_ * C_;
    __bf16* kT  = qT  + (size_t)B_ * L_ * CK_;
    __bf16* vW  = kT  + (size_t)B_ * L_ * CK_;
    __bf16* Wqb = vW  + (size_t)B_ * C_ * L_;
    __bf16* Wkb = Wqb + (size_t)CK_ * C_;
    __bf16* Wvb = Wkb + (size_t)CK_ * C_;

    k_cvt<<<dim3((CK_ * C_ + 255) / 256), 256, 0, stream>>>(Wq, Wqb, CK_ * C_);
    k_cvt<<<dim3((CK_ * C_ + 255) / 256), 256, 0, stream>>>(Wk, Wkb, CK_ * C_);
    k_cvt<<<dim3((C_  * C_ + 255) / 256), 256, 0, stream>>>(Wv, Wvb, C_ * C_);
    k_transpose<<<dim3(L_ / 32, C_ / 32, B_), dim3(32, 8), 0, stream>>>(x, xT);

    k_proj<<<dim3(L_ / 256, CK_ / 16, B_), 128, 0, stream>>>(Wqb, bq, xT, qT, nullptr);
    k_proj<<<dim3(L_ / 256, CK_ / 16, B_), 128, 0, stream>>>(Wkb, bk, xT, kT, nullptr);
    k_proj<<<dim3(L_ / 256, C_  / 16, B_), 128, 0, stream>>>(Wvb, bv, xT, nullptr, vW);

    k_attn<<<dim3(L_ / 32, B_), 256, 0, stream>>>(qT, kT, vW, x, gamma, out);
}
